// MinimalGSA_31817117729240
// MI455X (gfx1250) — matrix operand order repaired and checked by reading, not compiled, not run on hardware
//
#include <hip/hip_runtime.h>

// ---------------------------------------------------------------------------
// MinimalGSA on MI455X (gfx1250): bf16 WMMA for all GEMMs, fp32 accumulate.
// Shapes fixed by the reference: B=2, T=2048, D=1024, H=8, S=32, Dh=128.
// ---------------------------------------------------------------------------

#define B_   2
#define T_   2048
#define D_   1024
#define H_   8
#define S_   32
#define DH_  128
#define EPS_ 1e-6f

typedef __bf16 v16bf __attribute__((ext_vector_type(16)));
typedef float  v8f   __attribute__((ext_vector_type(8)));
typedef int    v4i   __attribute__((ext_vector_type(4)));

// Address-space pointer typedefs for the async-to-LDS builtin.
typedef __attribute__((address_space(1))) void* as1_void_t;   // global void*
typedef __attribute__((address_space(3))) void* as3_void_t;   // LDS void*
typedef __attribute__((address_space(1))) v4i*  as1_v4i_t;    // global int4*
typedef __attribute__((address_space(3))) v4i*  as3_v4i_t;    // LDS int4*

#if defined(__has_builtin)
#  if __has_builtin(__builtin_amdgcn_global_load_async_to_lds_b128)
#    define HAVE_ASYNC_LDS 1
#  endif
#  if __has_builtin(__builtin_amdgcn_s_wait_asynccnt)
#    define HAVE_WAIT_ASYNC 1
#  endif
#endif

// Hardware bf16 conversion (RNE on gfx1250).
static __device__ __forceinline__ unsigned short f32_to_bf16(float f) {
    union { __bf16 h; unsigned short u; } c;
    c.h = (__bf16)f;
    return c.u;
}

static __device__ __forceinline__ unsigned int pack_bf16x2(float a, float b) {
    union { __bf16 h[2]; unsigned int u; } c;
    c.h[0] = (__bf16)a;
    c.h[1] = (__bf16)b;
    return c.u;
}

union Frag16 {
    struct { uint4 lo, hi; } q;
    v16bf v;
};

// Build a 16-element bf16 fragment from two contiguous 8-element chunks.
static __device__ __forceinline__ v16bf make_frag(const unsigned short* lo8,
                                                  const unsigned short* hi8) {
    Frag16 f;
    f.q.lo = *(const uint4*)lo8;
    f.q.hi = *(const uint4*)hi8;
    return f.v;
}

// ---------------------------------------------------------------------------
// Generic C(M x N) = A(M x K) @ B(K x N) + bias, fp32 in/out, bf16 WMMA core.
// Workgroup tile 64x64, 4 waves, each wave 32x32 (2x2 of 16x16 WMMA tiles).
// M, N multiples of 64; K multiple of 32 (true for all uses here).
// A-frag layout (16-bit 16x32): lane m=lane&15, kg=lane>>4; elems 0..7 hold
//   K=kg*8+0..7, elems 8..15 hold K=kg*8+16..23  -> two 16B LDS reads.
// B-frag layout (16-bit 32x16): lane n=lane&15, kg=lane>>4; elems 0..7 hold
//   K=kg*8+0..7, elems 8..15 hold K=kg*8+16..23 -> store B transposed in LDS so K is contiguous.
// ---------------------------------------------------------------------------
__global__ void __launch_bounds__(128)
gemm_f32_bf16wmma(const float* __restrict__ A, int lda,
                  const float* __restrict__ Bm, int ldb,
                  const float* __restrict__ bias,
                  float* __restrict__ C, int ldc,
                  int K) {
    __shared__ unsigned short ldsA[64 * 32];   // [row][k]
    __shared__ unsigned short ldsB[64 * 32];   // transposed: [n][k]

    const int tid  = threadIdx.x;
    const int lane = tid & 31;
    const int wave = tid >> 5;
    const int rowBase = blockIdx.y * 64;
    const int colBase = blockIdx.x * 64;
    const int waveM = (wave >> 1) * 32;
    const int waveN = (wave & 1) * 32;

    const int mh = lane & 15;
    const int kg = lane >> 4;

    v8f acc00 = {}, acc01 = {}, acc10 = {}, acc11 = {};

    for (int kk = 0; kk < K; kk += 32) {
        if (kk + 32 < K) {
            __builtin_prefetch(&A[(size_t)(rowBase + (tid >> 1)) * lda + kk + 32], 0, 1);
            __builtin_prefetch(&Bm[(size_t)(kk + 32 + (tid >> 2)) * ldb + colBase], 0, 1);
        }
        // Stage A: 64 rows x 32 k; each thread moves 4 float4 quads.
        #pragma unroll
        for (int r = 0; r < 4; ++r) {
            int qi  = r * 128 + tid;           // 512 quads
            int row = qi >> 3;                 // 8 quads per row
            int c4  = (qi & 7) * 4;
            float4 f = *(const float4*)&A[(size_t)(rowBase + row) * lda + kk + c4];
            unsigned int* dst = (unsigned int*)&ldsA[row * 32 + c4];
            dst[0] = pack_bf16x2(f.x, f.y);
            dst[1] = pack_bf16x2(f.z, f.w);
        }
        // Stage B transposed: 32 k x 64 n; each thread moves 8 float2 pairs.
        #pragma unroll
        for (int r = 0; r < 8; ++r) {
            int pi = r * 128 + tid;            // 1024 pairs
            int k  = pi >> 5;                  // 32 pairs per k-row
            int n2 = (pi & 31) * 2;
            float2 f = *(const float2*)&Bm[(size_t)(kk + k) * ldb + colBase + n2];
            ldsB[(n2 + 0) * 32 + k] = f32_to_bf16(f.x);
            ldsB[(n2 + 1) * 32 + k] = f32_to_bf16(f.y);
        }
        __syncthreads();

        const unsigned short* a0p = &ldsA[(waveM + mh) * 32 + kg * 8];
        const unsigned short* a1p = &ldsA[(waveM + 16 + mh) * 32 + kg * 8];
        v16bf a0 = make_frag(a0p, a0p + 16);
        v16bf a1 = make_frag(a1p, a1p + 16);
        const unsigned short* b0p = &ldsB[(waveN + mh) * 32 + kg * 8];
        const unsigned short* b1p = &ldsB[(waveN + 16 + mh) * 32 + kg * 8];
        v16bf b0 = make_frag(b0p, b0p + 16);
        v16bf b1 = make_frag(b1p, b1p + 16);

        acc00 = __builtin_amdgcn_wmma_f32_16x16x32_bf16(false, a0, false, b0, (short)0, acc00, false, false);
        acc01 = __builtin_amdgcn_wmma_f32_16x16x32_bf16(false, a0, false, b1, (short)0, acc01, false, false);
        acc10 = __builtin_amdgcn_wmma_f32_16x16x32_bf16(false, a1, false, b0, (short)0, acc10, false, false);
        acc11 = __builtin_amdgcn_wmma_f32_16x16x32_bf16(false, a1, false, b1, (short)0, acc11, false, false);
        __syncthreads();
    }

    // C/D layout: VGPR r holds (m = r + 8*(lane>>4), n = lane&15)
    const int mOff = kg * 8;
    const int n0 = colBase + waveN + mh;
    const int n1 = n0 + 16;
    const float bias0 = bias ? bias[n0] : 0.0f;
    const float bias1 = bias ? bias[n1] : 0.0f;
    #pragma unroll
    for (int r = 0; r < 8; ++r) {
        int m0 = rowBase + waveM + mOff + r;
        int m1 = m0 + 16;
        C[(size_t)m0 * ldc + n0] = acc00[r] + bias0;
        C[(size_t)m0 * ldc + n1] = acc01[r] + bias1;
        C[(size_t)m1 * ldc + n0] = acc10[r] + bias0;
        C[(size_t)m1 * ldc + n1] = acc11[r] + bias1;
    }
}

// ---------------------------------------------------------------------------
// Splat weights: for each (b,t,h), 32 lanes = 32 splats.
//   qa(b,h,t,s) = exp(-0.5*(|q-c|/(exp(ls)+EPS))^2) * exp(la)   (bf16)
//   kw(b,h,t,s) = exp(-0.5*(|k-c|/(exp(ls)+EPS))^2)             (bf16)
// All row pointers are 512B-aligned -> float4 loads. __expf -> v_exp_f32.
// ---------------------------------------------------------------------------
__global__ void __launch_bounds__(256)
splat_weights_kernel(const float* __restrict__ qkv,
                     const float* __restrict__ centers,
                     const float* __restrict__ log_scales,
                     const float* __restrict__ log_amps,
                     unsigned short* __restrict__ qa,
                     unsigned short* __restrict__ kw) {
    const int lane = threadIdx.x & 31;
    const int wave = threadIdx.x >> 5;
    const int g = blockIdx.x * 8 + wave;        // in [0, B*T*H)
    const int h = g % H_;
    const int t = (g / H_) % T_;
    const int b = g / (H_ * T_);
    const int s = lane;

    const float* qp = qkv + (size_t)(b * T_ + t) * (3 * D_) + h * DH_;
    const float* kp = qp + D_;
    const float* cp = centers + ((size_t)h * S_ + s) * DH_;

    float dotq = 0.f, dotk = 0.f, qsq = 0.f, ksq = 0.f, csq = 0.f;
    #pragma unroll 4
    for (int d = 0; d < DH_; d += 4) {
        float4 c4 = *(const float4*)&cp[d];
        float4 q4 = *(const float4*)&qp[d];
        float4 k4 = *(const float4*)&kp[d];
        dotq += q4.x * c4.x + q4.y * c4.y + q4.z * c4.z + q4.w * c4.w;
        dotk += k4.x * c4.x + k4.y * c4.y + k4.z * c4.z + k4.w * c4.w;
        qsq  += q4.x * q4.x + q4.y * q4.y + q4.z * q4.z + q4.w * q4.w;
        ksq  += k4.x * k4.x + k4.y * k4.y + k4.z * k4.z + k4.w * k4.w;
        csq  += c4.x * c4.x + c4.y * c4.y + c4.z * c4.z + c4.w * c4.w;
    }
    float sc  = __expf(log_scales[h * S_ + s]) + EPS_;
    float amp = __expf(log_amps[h * S_ + s]);
    float dq  = sqrtf(fmaxf(qsq + csq - 2.f * dotq, 0.f));
    float dk  = sqrtf(fmaxf(ksq + csq - 2.f * dotk, 0.f));
    float rq  = dq / sc, rk = dk / sc;
    size_t base = ((size_t)(b * H_ + h) * T_ + t) * S_ + s;
    qa[base] = f32_to_bf16(__expf(-0.5f * rq * rq) * amp);
    kw[base] = f32_to_bf16(__expf(-0.5f * rk * rk));
}

// ---------------------------------------------------------------------------
// logits(b,h,i,j) = sum_s qa(b,h,i,s)*kw(b,h,j,s) / temperature
// K = S = 32 -> exactly one WMMA per 16x16 tile, fragments loaded straight
// from global (rows of length 32 bf16 are K-contiguous for both operands).
// 8 waves per block, one tile per wave.
// ---------------------------------------------------------------------------
__global__ void __launch_bounds__(256)
logits_wmma(const unsigned short* __restrict__ qa,   // (B*H, T, S)
            const unsigned short* __restrict__ kw,   // (B*H, T, S)
            const float* __restrict__ temperature,
            float* __restrict__ logits) {            // (B*H, T, T)
    const int lane = threadIdx.x & 31;
    const int wave = threadIdx.x >> 5;
    const int z = blockIdx.z;                        // b*H + h
    const int tile = blockIdx.x * 8 + wave;          // [0, 128*128)
    const int ti = tile >> 7;
    const int tj = tile & 127;
    const float invT = 1.0f / temperature[0];
    const int mh = lane & 15;
    const int kg = lane >> 4;

    const unsigned short* ap = qa + ((size_t)z * T_ + ti * 16 + mh) * S_ + kg * 8;
    const unsigned short* bp = kw + ((size_t)z * T_ + tj * 16 + mh) * S_ + kg * 8;
    v16bf a = make_frag(ap, ap + 16);
    v16bf b = make_frag(bp, bp + 16);

    v8f acc = {};
    acc = __builtin_amdgcn_wmma_f32_16x16x32_bf16(false, a, false, b, (short)0, acc, false, false);

    float* out = logits + (size_t)z * T_ * T_;
    const int i0 = ti * 16 + kg * 8;
    const int j  = tj * 16 + mh;
    #pragma unroll
    for (int r = 0; r < 8; ++r)
        out[(size_t)(i0 + r) * T_ + j] = acc[r] * invT;
}

// ---------------------------------------------------------------------------
// Row softmax over j (T=2048). One 256-thread block (8 wave32) per row.
// Writes fp32 attention to d_out (layout b,i,j,h) and bf16 copy for attn@V.
// __expf -> hardware v_exp_f32 (67M elements pass through here).
// ---------------------------------------------------------------------------
__global__ void __launch_bounds__(256)
softmax_kernel(const float* __restrict__ logits,      // (B*H, T, T)
               unsigned short* __restrict__ attnb,    // (B*H, T, T) bf16
               float* __restrict__ attn_out) {        // (B, T, T, H)
    const int row = blockIdx.x;                        // z*T + i
    const int z = row / T_;
    const int i = row % T_;
    const int b = z / H_;
    const int h = z % H_;
    const float* lr = logits + (size_t)row * T_;
    const int tid = threadIdx.x, lane = tid & 31, wave = tid >> 5;
    __shared__ float red[8];

    float vals[8];
    float mx = -3.402823466e38f;
    #pragma unroll
    for (int u = 0; u < 8; ++u) {
        vals[u] = lr[tid + u * 256];
        mx = fmaxf(mx, vals[u]);
    }
    #pragma unroll
    for (int off = 16; off; off >>= 1) mx = fmaxf(mx, __shfl_xor(mx, off, 32));
    if (lane == 0) red[wave] = mx;
    __syncthreads();
    float gmax = red[0];
    #pragma unroll
    for (int w = 1; w < 8; ++w) gmax = fmaxf(gmax, red[w]);
    __syncthreads();

    float s = 0.f;
    #pragma unroll
    for (int u = 0; u < 8; ++u) { vals[u] = __expf(vals[u] - gmax); s += vals[u]; }
    #pragma unroll
    for (int off = 16; off; off >>= 1) s += __shfl_xor(s, off, 32);
    if (lane == 0) red[wave] = s;
    __syncthreads();
    float gs = 0.f;
    #pragma unroll
    for (int w = 0; w < 8; ++w) gs += red[w];
    const float inv = 1.0f / gs;

    unsigned short* ab = attnb + (size_t)row * T_;
    float* ao = attn_out + ((size_t)(b * T_ + i)) * T_ * H_ + h;
    #pragma unroll
    for (int u = 0; u < 8; ++u) {
        int j = tid + u * 256;
        float p = vals[u] * inv;
        ab[j] = f32_to_bf16(p);
        ao[(size_t)j * H_] = p;
    }
}

// ---------------------------------------------------------------------------
// O(b,i,h,:) = sum_j attn(b,h,i,j) * V(b,j,h,:)  per (b,h): 2048x128 = A@V
// A already bf16 in memory: pure copy to LDS -> async-to-LDS (ASYNCcnt),
// drained with s_wait_asynccnt before the workgroup barrier publishes LDS.
// V staged from strided fp32 qkv.
// ---------------------------------------------------------------------------
__global__ void __launch_bounds__(128)
gemm_attnV(const unsigned short* __restrict__ attn,   // (B*H, T, T) bf16
           const float* __restrict__ qkv,             // (B, T, 3*D)
           float* __restrict__ O) {                   // (B, T, D)
    __shared__ unsigned short ldsA[64 * 32];
    __shared__ unsigned short ldsB[64 * 32];          // transposed [n][k]

    const int tid  = threadIdx.x;
    const int lane = tid & 31;
    const int wave = tid >> 5;
    const int z = blockIdx.z;
    const int b = z / H_, h = z % H_;
    const int rowBase = blockIdx.y * 64;              // i
    const int colBase = blockIdx.x * 64;              // d within Dh
    const unsigned short* Ab = attn + (size_t)z * T_ * T_;
    const float* Vb = qkv + (size_t)b * T_ * (3 * D_) + 2 * D_ + h * DH_;

    const int waveM = (wave >> 1) * 32;
    const int waveN = (wave & 1) * 32;
    const int mh = lane & 15;
    const int kg = lane >> 4;

    v8f acc00 = {}, acc01 = {}, acc10 = {}, acc11 = {};

    for (int kk = 0; kk < T_; kk += 32) {
        // Stage A (bf16 copy): 64 rows x 32 k = 4KB; 2 x B128 per thread.
        #pragma unroll
        for (int r = 0; r < 2; ++r) {
            int oi  = r * 128 + tid;           // 256 octs of 8 bf16
            int row = oi >> 2;                 // 4 octs per row
            int c8  = (oi & 3) * 8;
            const unsigned short* src = &Ab[(size_t)(rowBase + row) * T_ + kk + c8];
            unsigned short* dst = &ldsA[row * 32 + c8];
#ifdef HAVE_ASYNC_LDS
            __builtin_amdgcn_global_load_async_to_lds_b128(
                (as1_v4i_t)(as1_void_t)(void*)src,
                (as3_v4i_t)(as3_void_t)(void*)dst, 0, 0);
#else
            *(uint4*)dst = *(const uint4*)src;
#endif
        }
        // Stage B transposed: 32 k x 64 n from strided fp32 V.
        #pragma unroll
        for (int r = 0; r < 8; ++r) {
            int pi = r * 128 + tid;            // 1024 pairs
            int k  = pi >> 5;
            int n2 = (pi & 31) * 2;
            float2 f = *(const float2*)&Vb[(size_t)(kk + k) * (3 * D_) + colBase + n2];
            ldsB[(n2 + 0) * 32 + k] = f32_to_bf16(f.x);
            ldsB[(n2 + 1) * 32 + k] = f32_to_bf16(f.y);
        }
#ifdef HAVE_ASYNC_LDS
#  ifdef HAVE_WAIT_ASYNC
        __builtin_amdgcn_s_wait_asynccnt(0);
#  else
        asm volatile("s_wait_asynccnt 0" ::: "memory");
#  endif
#endif
        __syncthreads();

        const unsigned short* a0p = &ldsA[(waveM + mh) * 32 + kg * 8];
        const unsigned short* a1p = &ldsA[(waveM + 16 + mh) * 32 + kg * 8];
        v16bf a0 = make_frag(a0p, a0p + 16);
        v16bf a1 = make_frag(a1p, a1p + 16);
        const unsigned short* b0p = &ldsB[(waveN + mh) * 32 + kg * 8];
        const unsigned short* b1p = &ldsB[(waveN + 16 + mh) * 32 + kg * 8];
        v16bf b0 = make_frag(b0p, b0p + 16);
        v16bf b1 = make_frag(b1p, b1p + 16);

        acc00 = __builtin_amdgcn_wmma_f32_16x16x32_bf16(false, a0, false, b0, (short)0, acc00, false, false);
        acc01 = __builtin_amdgcn_wmma_f32_16x16x32_bf16(false, a0, false, b1, (short)0, acc01, false, false);
        acc10 = __builtin_amdgcn_wmma_f32_16x16x32_bf16(false, a1, false, b0, (short)0, acc10, false, false);
        acc11 = __builtin_amdgcn_wmma_f32_16x16x32_bf16(false, a1, false, b1, (short)0, acc11, false, false);
        __syncthreads();
    }

    const int mOff = kg * 8;
    const int n0 = colBase + waveN + mh;
    const int n1 = n0 + 16;
    #pragma unroll
    for (int r = 0; r < 8; ++r) {
        int m0 = rowBase + waveM + mOff + r;
        int m1 = m0 + 16;
        O[(size_t)(b * T_ + m0) * D_ + h * DH_ + n0] = acc00[r];
        O[(size_t)(b * T_ + m0) * D_ + h * DH_ + n1] = acc01[r];
        O[(size_t)(b * T_ + m1) * D_ + h * DH_ + n0] = acc10[r];
        O[(size_t)(b * T_ + m1) * D_ + h * DH_ + n1] = acc11[r];
    }
}

// ---------------------------------------------------------------------------
// Launch
// ---------------------------------------------------------------------------
extern "C" void kernel_launch(void* const* d_in, const int* in_sizes, int n_in,
                              void* d_out, int out_size, void* d_ws, size_t ws_size,
                              hipStream_t stream) {
    const float* x       = (const float*)d_in[0];
    const float* Wqkv    = (const float*)d_in[1];
    const float* bqkv    = (const float*)d_in[2];
    const float* Wout    = (const float*)d_in[3];
    const float* bout    = (const float*)d_in[4];
    const float* centers = (const float*)d_in[5];
    const float* lsc     = (const float*)d_in[6];
    const float* lam     = (const float*)d_in[7];
    const float* temp    = (const float*)d_in[8];

    char* ws = (char*)d_ws;
    size_t off = 0;
    auto take = [&](size_t bytes) -> char* {
        char* p = ws + off;
        off = (off + bytes + 255) & ~(size_t)255;
        return p;
    };
    float*          qkv    = (float*)take((size_t)B_ * T_ * 3 * D_ * sizeof(float));
    unsigned short* qa     = (unsigned short*)take((size_t)B_ * H_ * T_ * S_ * 2);
    unsigned short* kw     = (unsigned short*)take((size_t)B_ * H_ * T_ * S_ * 2);
    float*          logits = (float*)take((size_t)B_ * H_ * T_ * T_ * sizeof(float));
    unsigned short* attnb  = (unsigned short*)take((size_t)B_ * H_ * T_ * T_ * 2);
    float*          Obuf   = (float*)take((size_t)B_ * T_ * D_ * sizeof(float));

    float* out_proj = (float*)d_out;                         // (B,T,D)
    float* attn_out = out_proj + (size_t)B_ * T_ * D_;       // (B,T,T,H)

    // 1) qkv = x @ Wqkv + bqkv : (4096x1024)@(1024x3072)
    gemm_f32_bf16wmma<<<dim3(3 * D_ / 64, B_ * T_ / 64), 128, 0, stream>>>(
        x, D_, Wqkv, 3 * D_, bqkv, qkv, 3 * D_, D_);

    // 2) splat weights -> qa, kw (bf16, (B*H,T,S))
    splat_weights_kernel<<<dim3(B_ * T_ * H_ / 8), 256, 0, stream>>>(
        qkv, centers, lsc, lam, qa, kw);

    // 3) logits = qa @ kw^T / temperature  (one WMMA per 16x16 tile)
    logits_wmma<<<dim3((T_ / 16) * (T_ / 16) / 8, 1, B_ * H_), 256, 0, stream>>>(
        qa, kw, temp, logits);

    // 4) softmax rows -> fp32 attention output + bf16 copy
    softmax_kernel<<<dim3(B_ * H_ * T_), 256, 0, stream>>>(logits, attnb, attn_out);

    // 5) O = attn @ V  per (b,h)
    gemm_attnV<<<dim3(DH_ / 64, T_ / 64, B_ * H_), 128, 0, stream>>>(attnb, qkv, Obuf);

    // 6) out = O @ Wout + bout : (4096x1024)@(1024x1024)
    gemm_f32_bf16wmma<<<dim3(D_ / 64, B_ * T_ / 64), 128, 0, stream>>>(
        Obuf, D_, Wout, D_, bout, out_proj, D_, D_);
}
